// EngramLayer_51625506897945
// MI455X (gfx1250) — compile-verified, matrix operand order checked by reading, not run on hardware
//
#include <hip/hip_runtime.h>
#include <hip/hip_bf16.h>

// ---------------- problem constants ----------------
#define BATCH   4
#define SEQLEN  4096
#define NTOK    (BATCH * SEQLEN)        // 16384  (GEMM M)
#define HDIM    2560                    // hidden dim
#define EDIM    1024                    // embedding dim (GEMM K)
#define NPROJ   (2 * HDIM)              // 5120  (GEMM N: [v ; k])
#define DHEAD   128
#define KSIZE   4
#define EPS     1e-6f

__device__ __constant__ long long kOff[8] = {
    0LL, 250007LL, 500020LL, 750047LL, 1000078LL, 1250115LL, 1500158LL, 1750207LL
};

typedef __attribute__((ext_vector_type(16))) _Float16 v16h;
typedef __attribute__((ext_vector_type(8)))  float    v8f;

union FragH { uint4 u[2]; v16h v; };

static __device__ __forceinline__ _Float16 f2h(float f) { return (_Float16)f; }

// ---------------- kernel 1: hash gather + f32 -> f16 ----------------
// one token per block, 256 threads; e_h is [NTOK, EDIM] f16 row-major
__global__ void gather_embed(const long long* __restrict__ idx,
                             const float* __restrict__ table,
                             _Float16* __restrict__ e_h) {
    const int token = blockIdx.x;
    const long long* ip = idx + (size_t)token * 8;
    for (int i = threadIdx.x; i < EDIM; i += 256) {
        const int h = i >> 7;          // head
        const int d = i & 127;         // dim within head
        const long long row = ip[h] + kOff[h];
        e_h[(size_t)token * EDIM + i] = f2h(table[row * DHEAD + d]);
    }
}

// ---------------- kernel 2: weight f32 -> f16, [w_v ; w_k] concat ----------------
__global__ void cvt_weights(const float* __restrict__ wv,
                            const float* __restrict__ wk,
                            _Float16* __restrict__ wh) {
    const size_t half = (size_t)HDIM * EDIM;           // 2621440
    size_t i = (size_t)blockIdx.x * 256 + threadIdx.x;
    if (i < half)            wh[i] = f2h(wv[i]);
    else if (i < 2 * half)   wh[i] = f2h(wk[i - half]);
}

// ---------------- kernel 3: fused GEMM  C[M,N] = A[M,K] * W[N,K]^T ----------------
// grid = (N/128, M/128), block = 256 (8 waves, arranged 4(M) x 2(N)).
// Each wave computes a 32x64 tile: 2 M sub-tiles x 4 N sub-tiles = 8 WMMAs / K-step,
// sharing the 4 B fragments across both M sub-tiles (12 b128 loads per 8 WMMAs).
__global__ void gemm_wmma(const _Float16* __restrict__ A,
                          const _Float16* __restrict__ W,
                          float* __restrict__ C) {
    const int wave = threadIdx.x >> 5;
    const int lane = threadIdx.x & 31;
    const int half = lane >> 4;        // 0: K {0..7,16..23}; 1: K {8..15,24..31}
    const int r    = lane & 15;
    const int m0   = blockIdx.y * 128 + (wave >> 1) * 32;
    const int n0   = blockIdx.x * 128 + (wave & 1) * 64;

    v8f acc[2][4];
    #pragma unroll
    for (int mi = 0; mi < 2; ++mi)
        #pragma unroll
        for (int t = 0; t < 4; ++t) acc[mi][t] = (v8f){};

    const _Float16* arow[2];
    arow[0] = A + (size_t)(m0 +  0 + r) * EDIM;
    arow[1] = A + (size_t)(m0 + 16 + r) * EDIM;
    const _Float16* wrow[4];
    #pragma unroll
    for (int t = 0; t < 4; ++t) wrow[t] = W + (size_t)(n0 + t * 16 + r) * EDIM;

    #pragma unroll 2
    for (int k = 0; k < EDIM; k += 32) {
        const int kb = k + half * 8;

        FragH af[2];
        #pragma unroll
        for (int mi = 0; mi < 2; ++mi) {
            const uint4* ap = (const uint4*)(arow[mi] + kb);
            af[mi].u[0] = ap[0];
            af[mi].u[1] = ap[2];               // +16 halves = +2 uint4
        }
        __builtin_prefetch(arow[0] + kb + 32, 0, 1);   // global_prefetch next K slab
        __builtin_prefetch(arow[1] + kb + 32, 0, 1);

        #pragma unroll
        for (int t = 0; t < 4; ++t) {
            FragH bf;
            const uint4* bp = (const uint4*)(wrow[t] + kb);
            bf.u[0] = bp[0];
            bf.u[1] = bp[2];
            acc[0][t] = __builtin_amdgcn_wmma_f32_16x16x32_f16(
                false, af[0].v, false, bf.v, (short)0, acc[0][t], false, false);
            acc[1][t] = __builtin_amdgcn_wmma_f32_16x16x32_f16(
                false, af[1].v, false, bf.v, (short)0, acc[1][t], false, false);
        }
    }

    // C/D layout: VGPR j -> row m0 + mi*16 + half*8 + j, column n0 + t*16 + r
    #pragma unroll
    for (int mi = 0; mi < 2; ++mi) {
        float* crow = C + (size_t)(m0 + mi * 16 + half * 8) * NPROJ + n0 + r;
        #pragma unroll
        for (int j = 0; j < 8; ++j) {
            float* cr = crow + (size_t)j * NPROJ;
            cr[ 0] = acc[mi][0][j];
            cr[16] = acc[mi][1][j];
            cr[32] = acc[mi][2][j];
            cr[48] = acc[mi][3][j];
        }
    }
}

// ---------------- block reduction (wave32 shuffle + LDS) ----------------
static __device__ __forceinline__ float block_sum(float v, volatile float* s8) {
    #pragma unroll
    for (int o = 16; o > 0; o >>= 1) v += __shfl_xor(v, o, 32);
    const int wave = threadIdx.x >> 5;
    const int lane = threadIdx.x & 31;
    if (lane == 0) s8[wave] = v;
    __syncthreads();
    float t = 0.f;
    #pragma unroll
    for (int i = 0; i < 8; ++i) t += s8[i];
    return t;
}

// ---------------- kernel 4: gating; in-place v_t -> v_tilde, k_t -> v_n ----------------
// one token per block, 256 threads, 10 elements per thread
__global__ void gate(const float* __restrict__ hs,
                     float* __restrict__ vk,          // [NTOK, NPROJ]; cols 0..H-1 = v, H..2H-1 = k
                     const float* __restrict__ g_h,
                     const float* __restrict__ g_k,
                     const float* __restrict__ g_v) {
    __shared__ float red[4][8];
    const int token = blockIdx.x;
    const float* hrow = hs + (size_t)token * HDIM;
    float* vrow = vk + (size_t)token * NPROJ;         // v part
    float* krow = vrow + HDIM;                        // k part

    float hv[10], kv[10], vv[10];
    float ssh = 0.f, ssk = 0.f;
    #pragma unroll
    for (int j = 0; j < 10; ++j) {
        const int i = threadIdx.x + j * 256;
        hv[j] = hrow[i];  kv[j] = krow[i];  vv[j] = vrow[i];
        ssh += hv[j] * hv[j];
        ssk += kv[j] * kv[j];
    }
    ssh = block_sum(ssh, red[0]);
    ssk = block_sum(ssk, red[1]);
    const float rh = rsqrtf(ssh / (float)HDIM + EPS);
    const float rk = rsqrtf(ssk / (float)HDIM + EPS);

    float dp = 0.f;
    #pragma unroll
    for (int j = 0; j < 10; ++j) {
        const int i = threadIdx.x + j * 256;
        dp += (hv[j] * rh * g_h[i]) * (kv[j] * rk * g_k[i]);
    }
    dp = block_sum(dp, red[2]) / sqrtf((float)HDIM);

    const float sgn   = (dp > 0.f) ? 1.f : ((dp < 0.f) ? -1.f : 0.f);
    const float stab  = sqrtf(fmaxf(fabsf(dp), 1e-6f)) * sgn;
    const float alpha = 1.f / (1.f + expf(-stab));

    float ssv = 0.f;
    #pragma unroll
    for (int j = 0; j < 10; ++j) {
        vv[j] *= alpha;                               // v_tilde
        ssv += vv[j] * vv[j];
    }
    ssv = block_sum(ssv, red[3]);
    const float rv = rsqrtf(ssv / (float)HDIM + EPS);

    #pragma unroll
    for (int j = 0; j < 10; ++j) {
        const int i = threadIdx.x + j * 256;
        vrow[i] = vv[j];                              // v_tilde (over v_t)
        krow[i] = vv[j] * rv * g_v[i];                // v_n     (over k_t)
    }
}

// ---------------- kernel 5: causal depthwise conv + SiLU + residual ----------------
__global__ void conv_silu_res(const float* __restrict__ hs,
                              const float* __restrict__ vk,   // v_tilde @ col 0, v_n @ col HDIM
                              const float* __restrict__ cw,   // [HDIM,4]
                              const float* __restrict__ cb,
                              float* __restrict__ out) {
    const size_t total = (size_t)NTOK * HDIM;
    const size_t i = (size_t)blockIdx.x * 256 + threadIdx.x;
    if (i >= total) return;
    const int h  = (int)(i % HDIM);
    const int tl = (int)(i / HDIM);                   // flat token
    const int l  = tl & (SEQLEN - 1);                 // position in sequence

    const float vt = vk[(size_t)tl * NPROJ + h];
    float acc = cb[h];
    #pragma unroll
    for (int j = 0; j < KSIZE; ++j) {
        const int ll = l - (KSIZE - 1) + j;
        if (ll >= 0)
            acc += cw[h * KSIZE + j] * vk[(size_t)(tl - (KSIZE - 1) + j) * NPROJ + HDIM + h];
    }
    const float silu = acc / (1.f + expf(-acc));
    out[i] = hs[i] + silu + vt;
}

// ---------------- launcher ----------------
extern "C" void kernel_launch(void* const* d_in, const int* in_sizes, int n_in,
                              void* d_out, int out_size, void* d_ws, size_t ws_size,
                              hipStream_t stream) {
    const float*     hs    = (const float*)    d_in[0];
    const long long* idx   = (const long long*)d_in[1];
    const float*     table = (const float*)    d_in[2];
    const float*     wv    = (const float*)    d_in[3];
    const float*     wk    = (const float*)    d_in[4];
    const float*     g_h   = (const float*)    d_in[5];
    const float*     g_k   = (const float*)    d_in[6];
    const float*     g_v   = (const float*)    d_in[7];
    const float*     cw    = (const float*)    d_in[8];
    const float*     cb    = (const float*)    d_in[9];
    float*           out   = (float*)d_out;

    // workspace layout
    char* ws = (char*)d_ws;
    _Float16* e_h = (_Float16*)ws;                                   // 32 MB
    _Float16* w_h = (_Float16*)(ws + (size_t)NTOK * EDIM * 2);       // 10 MB
    float*    vk  = (float*)   (ws + (size_t)NTOK * EDIM * 2
                                   + (size_t)NPROJ * EDIM * 2);      // 320 MB

    // 1) gather + downconvert activations
    gather_embed<<<NTOK, 256, 0, stream>>>(idx, table, e_h);

    // 2) downconvert [w_v ; w_k] weights
    {
        const size_t n = (size_t)NPROJ * EDIM;
        cvt_weights<<<(unsigned)((n + 255) / 256), 256, 0, stream>>>(wv, wk, w_h);
    }

    // 3) fused v/k projection GEMM: [16384 x 1024] x [1024 x 5120]
    {
        dim3 grid(NPROJ / 128, NTOK / 128);
        gemm_wmma<<<grid, 256, 0, stream>>>(e_h, w_h, vk);
    }

    // 4) context-aware gating (in-place over GEMM output)
    gate<<<NTOK, 256, 0, stream>>>(hs, vk, g_h, g_k, g_v);

    // 5) causal depthwise conv + SiLU + residual
    {
        const size_t n = (size_t)NTOK * HDIM;
        conv_silu_res<<<(unsigned)((n + 255) / 256), 256, 0, stream>>>(hs, vk, cw, cb, out);
    }
}
